// ARMHGNN_56100862821052
// MI455X (gfx1250) — compile-verified, matrix operand order checked by reading, not run on hardware
//
#include <hip/hip_runtime.h>
#include <hip/hip_bf16.h>

// ---------------------------------------------------------------------------
// ARMHGNN (HetGNN-style) forward on MI455X / gfx1250.
//
// All BiLSTM gate GEMMs run on v_wmma_f32_16x16x32_f16:
//   gates[16,256] = [x_t | h] (16x192, f16) x Wcat^T (192x256, f16), f32 accum
// One workgroup (256 threads = 8 wave32) processes one 16-row tile; weights
// are packed to LDS once per direction and held as register B-fragments
// across the recurrent scan. A-fragments for all 6 K-steps are loaded into
// distinct registers up front so the ds_load_b128s clause together and the
// 12 WMMAs issue behind a single s_wait_dscnt. Elementwise cell update on VALU.
// ---------------------------------------------------------------------------

typedef _Float16 v8h  __attribute__((ext_vector_type(8)));
typedef _Float16 v16h __attribute__((ext_vector_type(16)));
typedef float    v8f  __attribute__((ext_vector_type(8)));

struct Gather {
    const float* base[6];   // per-timestep base pointer (column offset folded in)
    long         stride[6]; // per-timestep row stride in floats
};

__device__ __forceinline__ float sigf(float x) { return 1.0f / (1.0f + expf(-x)); }

// Build a 16x32 f16 A/B fragment from a contiguous-K LDS row.
// Per ISA 7.12.2 (16-bit A 16x32): lane L holds M(or N)=L&15; half idx i maps to
// K = k0 + ((i>=8)?16:0) + ((L&16)?8:0) + (i&7)  -> two ds_load_b128.
__device__ __forceinline__ v16h lds_frag(const _Float16* p /* = row + k0 + kbase */) {
    v8h lo = *(const v8h*)p;
    v8h hi = *(const v8h*)(p + 16);
    v16h r;
#pragma unroll
    for (int i = 0; i < 8; ++i) { r[i] = lo[i]; r[8 + i] = hi[i]; }
    return r;
}

// LDS layout (bytes, dynamic):
//   wlds  [256][192] f16   :      0 ..  98304
//   gates [16][256]  f32   :  98304 .. 114688
//   cst   [16][64]   f32   : 114688 .. 118784
//   hmean [16][128]  f32   : 118784 .. 126976
//   hcur  [16][64]   f16   : 126976 .. 129024
//   xall  [T][16][128] f16 : 129024 .. 129024 + T*4096
__global__ __launch_bounds__(256, 1)
void bilstm_mean_kernel(Gather g, const int* ids, int n_nb, const float* neigh_src,
                        const float* Wih_f, const float* Whh_f, const float* b_f,
                        const float* Wih_b, const float* Whh_b, const float* b_b,
                        float* out, int nRows, int T)
{
    extern __shared__ char smem[];
    _Float16* wlds  = (_Float16*)(smem);
    float*    gates = (float*)(smem + 98304);
    float*    cst   = (float*)(smem + 114688);
    float*    hmean = (float*)(smem + 118784);
    _Float16* hcur  = (_Float16*)(smem + 126976);
    _Float16* xall  = (_Float16*)(smem + 129024);

    const int tid  = threadIdx.x;
    const int wave = tid >> 5;
    const int lane = tid & 31;
    const int row0 = blockIdx.x * 16;

    // ---- gather sequence tile [T][16][128] into LDS, f32 -> f16 ----
    const int nChunk = T * 16 * 32;            // float4 chunks
    for (int idx = tid; idx < nChunk; idx += 256) {
        int c4 = idx & 31;
        int r  = (idx >> 5) & 15;
        int t  = idx >> 9;
        int row = row0 + r;
        if (row >= nRows) row = nRows - 1;     // clamp (tiles are exact anyway)
        const float* src;
        if (n_nb > 0) {
            // x[t][b] = content[b * n_nb + t]
            src = neigh_src + ((long)row * n_nb + t) * 128;
        } else {
            int id = ids ? ids[row] : row;
            src = g.base[t] + (long)id * g.stride[t];
        }
        float4 v = ((const float4*)src)[c4];
        _Float16* dst = xall + ((t * 16 + r) << 7) + (c4 << 2);
        dst[0] = (_Float16)v.x; dst[1] = (_Float16)v.y;
        dst[2] = (_Float16)v.z; dst[3] = (_Float16)v.w;
    }
    for (int e = tid; e < 2048; e += 256) hmean[e] = 0.0f;
    __syncthreads();

    for (int dir = 0; dir < 2; ++dir) {
        const float* Wih  = dir ? Wih_b : Wih_f;
        const float* Whh  = dir ? Whh_b : Whh_f;
        const float* bias = dir ? b_b   : b_f;

        // ---- pack Wcat = [Wih | Whh] row n=tid into LDS as f16 [256][192] ----
        {
            const float* wi = Wih + tid * 128;
            const float* wh = Whh + tid * 64;
            _Float16* wr = wlds + tid * 192;
#pragma unroll 4
            for (int k = 0; k < 128; ++k) wr[k] = (_Float16)wi[k];
#pragma unroll 4
            for (int k = 0; k < 64; ++k)  wr[128 + k] = (_Float16)wh[k];
        }
        for (int e = tid; e < 1024; e += 256) { cst[e] = 0.0f; hcur[e] = (_Float16)0.0f; }
        __syncthreads();

        // ---- preload register-resident B fragments: 2 N-tiles x 6 K-steps ----
        v16h bfrag[2][6];
        {
            const int kb = (lane & 16) ? 8 : 0;
#pragma unroll
            for (int tt = 0; tt < 2; ++tt) {
                int n = (wave << 5) + (tt << 4) + (lane & 15);
#pragma unroll
                for (int kt = 0; kt < 6; ++kt)
                    bfrag[tt][kt] = lds_frag(wlds + n * 192 + kt * 32 + kb);
            }
        }

        // ---- recurrent scan ----
        for (int s = 0; s < T; ++s) {
            const int t = dir ? (T - 1 - s) : s;

            const int m  = lane & 15;
            const int kb = (lane & 16) ? 8 : 0;

            // Load ALL six A fragments into distinct registers first so the
            // ds_load_b128s issue back-to-back and the WMMA chain runs behind
            // a single s_wait_dscnt.
            v16h afrag[6];
#pragma unroll
            for (int kt = 0; kt < 4; ++kt)
                afrag[kt] = lds_frag(xall + ((t * 16 + m) << 7) + kt * 32 + kb);
#pragma unroll
            for (int kt = 4; kt < 6; ++kt)
                afrag[kt] = lds_frag(hcur + (m << 6) + (kt - 4) * 32 + kb);

            v8f acc0 = {}; v8f acc1 = {};
#pragma unroll
            for (int kt = 0; kt < 6; ++kt) {
                acc0 = __builtin_amdgcn_wmma_f32_16x16x32_f16(
                           false, afrag[kt], false, bfrag[0][kt], (short)0, acc0, false, false);
                acc1 = __builtin_amdgcn_wmma_f32_16x16x32_f16(
                           false, afrag[kt], false, bfrag[1][kt], (short)0, acc1, false, false);
            }
            // scatter D tiles to gates LDS (C/D layout: lane holds col N=lane&15,
            // rows r + 8*(lane>=16))
            {
                const int mh = (lane & 16) ? 8 : 0;
                const int n0 = (wave << 5) + (lane & 15);
#pragma unroll
                for (int r = 0; r < 8; ++r) {
                    gates[(mh + r) * 256 + n0]      = acc0[r];
                    gates[(mh + r) * 256 + n0 + 16] = acc1[r];
                }
            }
            __syncthreads();

            // ---- LSTM cell update (gate order i,f,g,o) ----
            for (int e = tid; e < 1024; e += 256) {
                int mm = e >> 6, j = e & 63;
                const float* gr = gates + mm * 256;
                float iv = sigf (gr[j]        + bias[j]);
                float fv = sigf (gr[64 + j]   + bias[64 + j]);
                float gv = tanhf(gr[128 + j]  + bias[128 + j]);
                float ov = sigf (gr[192 + j]  + bias[192 + j]);
                float c  = fv * cst[e] + iv * gv;
                cst[e] = c;
                float h = ov * tanhf(c);
                hcur[e] = (_Float16)h;
                hmean[(mm << 7) + (dir << 6) + j] += h;   // concat(fwd,bwd) mean accum
            }
            __syncthreads();
        }
        __syncthreads();   // protect wlds before next direction overwrites it
    }

    // ---- write mean over time: out[row, 0:128] ----
    const float invT = 1.0f / (float)T;
    for (int e = tid; e < 2048; e += 256) {
        int mm = e >> 7, col = e & 127;
        int row = row0 + mm;
        if (row < nRows) out[(long)row * 128 + col] = hmean[e] * invT;
    }
}

// ---- heterogeneous attention combine: leaky_relu + softmax over 4 sources ----
__global__ __launch_bounds__(128)
void att_combine_kernel(const float* cself, const float* s1, const float* s2,
                        const float* s3, const float* att, float* out, int B)
{
    __shared__ float red[4][128];
    __shared__ float wsh[4];
    const int row = blockIdx.x;
    const int j   = threadIdx.x;

    float q  = cself[row * 128 + j];
    float v0 = q;
    float v1 = s1[row * 128 + j];
    float v2 = s2[row * 128 + j];
    float v3 = s3[row * 128 + j];
    float a0 = att[j], a1 = att[128 + j];
    red[0][j] = a0 * q + a1 * v0;
    red[1][j] = a0 * q + a1 * v1;
    red[2][j] = a0 * q + a1 * v2;
    red[3][j] = a0 * q + a1 * v3;
    __syncthreads();
    for (int st = 64; st > 0; st >>= 1) {
        if (j < st) {
            red[0][j] += red[0][j + st]; red[1][j] += red[1][j + st];
            red[2][j] += red[2][j + st]; red[3][j] += red[3][j + st];
        }
        __syncthreads();
    }
    if (j == 0) {
        float sc[4], mx = -1e30f;
#pragma unroll
        for (int s = 0; s < 4; ++s) {
            float x = red[s][0];
            sc[s] = (x > 0.0f) ? x : 0.01f * x;   // leaky_relu, slope 0.01
            mx = fmaxf(mx, sc[s]);
        }
        float sum = 0.0f;
#pragma unroll
        for (int s = 0; s < 4; ++s) { sc[s] = expf(sc[s] - mx); sum += sc[s]; }
#pragma unroll
        for (int s = 0; s < 4; ++s) wsh[s] = sc[s] / sum;
    }
    __syncthreads();
    out[row * 128 + j] = wsh[0] * v0 + wsh[1] * v1 + wsh[2] * v2 + wsh[3] * v3;
}

// ---------------------------------------------------------------------------
extern "C" void kernel_launch(void* const* d_in, const int* in_sizes, int n_in,
                              void* d_out, int out_size, void* d_ws, size_t ws_size,
                              hipStream_t stream)
{
    (void)in_sizes; (void)n_in; (void)out_size; (void)ws_size;

    // Input order = setup_inputs() dict order, dicts flattened in insertion order.
    const float* p_t     = (const float*)d_in[0];
    const float* p_v_net = (const float*)d_in[1];
    const float* p_a_net = (const float*)d_in[2];
    const float* p_net   = (const float*)d_in[3];
    const float* a_coop  = (const float*)d_in[4];
    const float* a_net   = (const float*)d_in[5];
    const float* a_text  = (const float*)d_in[6];   // [A_N, 3*128]
    const float* v_net   = (const float*)d_in[7];
    const float* v_text  = (const float*)d_in[8];   // [V_N, 5*128]

    // rnn param groups: {Wih_f, Whh_f, b_f, Wih_b, Whh_b, b_b}
    const float* R[6][6];
    for (int gix = 0; gix < 6; ++gix)
        for (int k = 0; k < 6; ++k)
            R[gix][k] = (const float*)d_in[9 + gix * 6 + k];
    // groups: 0=a_content 1=p_content 2=v_content 3=a_neigh 4=p_neigh 5=v_neigh

    const float* att         = (const float*)d_in[45];  // [256,1]
    const int*   id_batch    = (const int*)d_in[46];    // [2000]
    const int*   a_neigh_ids = (const int*)d_in[47];    // [2000,10]
    const int*   p_neigh_ids = (const int*)d_in[48];    // [2000,20]
    const int*   v_neigh_ids = (const int*)d_in[49];    // [2000,3]

    const int B = 2000, A_NB = 10, P_NB = 20, V_NB = 3;

    // workspace carve (floats)
    float* ws        = (float*)d_ws;
    float* cont_self = ws;                          // [2000,128]
    float* cont_a    = cont_self + 2000 * 128;      // [20000,128]
    float* cont_p    = cont_a + 20000 * 128;        // [40000,128]
    float* cont_v    = cont_p + 40000 * 128;        // [6000,128]
    float* agg_a     = cont_v + 6000 * 128;         // [2000,128]
    float* agg_p     = agg_a + 2000 * 128;
    float* agg_v     = agg_p + 2000 * 128;

    // gather descriptors (column offsets folded into base pointers)
    Gather ga = {}, gp = {}, gv = {}, gn = {};
    ga.base[0] = a_coop;        ga.stride[0] = 128;
    ga.base[1] = a_net;         ga.stride[1] = 128;
    ga.base[2] = a_text;        ga.stride[2] = 384;
    ga.base[3] = a_text + 128;  ga.stride[3] = 384;
    ga.base[4] = a_text + 256;  ga.stride[4] = 384;

    gp.base[0] = p_t;           gp.stride[0] = 128;
    gp.base[1] = p_v_net;       gp.stride[1] = 128;
    gp.base[2] = p_a_net;       gp.stride[2] = 128;
    gp.base[3] = p_net;         gp.stride[3] = 128;

    gv.base[0] = v_net;         gv.stride[0] = 128;
    for (int t = 1; t < 6; ++t) { gv.base[t] = v_text + (t - 1) * 128; gv.stride[t] = 640; }

    auto smem = [](int T) { return (size_t)129024 + (size_t)T * 4096; };

#define LSTM_ARGS(gix) R[gix][0], R[gix][1], R[gix][2], R[gix][3], R[gix][4], R[gix][5]

    // Phase 1: content aggregations (BiLSTM over feature sequences)
    bilstm_mean_kernel<<<B / 16, 256, smem(5), stream>>>(
        ga, id_batch, 0, nullptr, LSTM_ARGS(0), cont_self, B, 5);
    bilstm_mean_kernel<<<(B * A_NB) / 16, 256, smem(5), stream>>>(
        ga, a_neigh_ids, 0, nullptr, LSTM_ARGS(0), cont_a, B * A_NB, 5);
    bilstm_mean_kernel<<<(B * P_NB) / 16, 256, smem(4), stream>>>(
        gp, p_neigh_ids, 0, nullptr, LSTM_ARGS(1), cont_p, B * P_NB, 4);
    bilstm_mean_kernel<<<(B * V_NB) / 16, 256, smem(6), stream>>>(
        gv, v_neigh_ids, 0, nullptr, LSTM_ARGS(2), cont_v, B * V_NB, 6);

    // Phase 2: neighbor aggregations (BiLSTM over the n_nb axis)
    bilstm_mean_kernel<<<B / 16, 256, smem(A_NB), stream>>>(
        gn, nullptr, A_NB, cont_a, LSTM_ARGS(3), agg_a, B, A_NB);
    bilstm_mean_kernel<<<B / 16, 256, smem(P_NB), stream>>>(
        gn, nullptr, P_NB, cont_p, LSTM_ARGS(4), agg_p, B, P_NB);
    bilstm_mean_kernel<<<B / 16, 256, smem(V_NB), stream>>>(
        gn, nullptr, V_NB, cont_v, LSTM_ARGS(5), agg_v, B, V_NB);

    // Phase 3: heterogeneous attention combine
    att_combine_kernel<<<B, 128, 0, stream>>>(
        cont_self, agg_a, agg_p, agg_v, att, (float*)d_out, B);

#undef LSTM_ARGS
}
